// GCNCheb_8340826489217
// MI455X (gfx1250) — compile-verified, matrix operand order checked by reading, not run on hardware
//
#include <hip/hip_runtime.h>

// ---------------------------------------------------------------------------
// Chebyshev GCN for MI455X (gfx1250, wave32, WMMA bf16 16x16x32)
// B=4, N=8192, F_IN=32, F_OUT=64, K=4
//
// P1 = L x ; P2 = L P1 ; P3 = L P2        (bf16 WMMA, f32 accum)
// out = x(W0-W2) + P1(W1-W3) + 2 P2 W2 + 2 P3 W3 + bias
//
// All activation operands stored transposed [col][node] in bf16 so both WMMA
// operands load with plain per-lane global_load_b128.
// ---------------------------------------------------------------------------

#define NN 8192      // nodes
#define NC 128       // stacked columns = B * F_IN
#define FOUT 64
#define NBATCH 4

typedef __attribute__((ext_vector_type(16))) __bf16 v16bf;
typedef __attribute__((ext_vector_type(8)))  float  v8f;

union V16 { v16bf v; uint4 q[2]; };

static __device__ __forceinline__ unsigned short f2bf(float f) {
    unsigned int u = __float_as_uint(f);
    u = (u + 0x7FFFu + ((u >> 16) & 1u)) >> 16;   // round-to-nearest-even
    return (unsigned short)u;
}

// Load one WMMA 16-bit operand (A or B side): lane r=lane&15 picks the
// row/col, kh = (lane>=16)*8 picks the K phase; two 16B chunks at +0 / +16.
static __device__ __forceinline__ V16 load_op(const unsigned short* p) {
    V16 t;
    t.q[0] = *reinterpret_cast<const uint4*>(p);       // K = kh .. kh+7
    t.q[1] = *reinterpret_cast<const uint4*>(p + 16);  // K = kh+16 .. kh+23
    return t;
}

static __device__ __forceinline__ v8f wmma_bf16(const V16& a, const V16& b, v8f c) {
    return __builtin_amdgcn_wmma_f32_16x16x32_bf16(
        false, a.v, false, b.v, (short)0, c, false, false);
}

// ------------------------- conversion kernels ------------------------------

// L fp32 [8192][8192] -> bf16, 8 elems / thread
__global__ __launch_bounds__(256) void conv_L(const float* __restrict__ L,
                                              unsigned short* __restrict__ Lb) {
    size_t i = ((size_t)blockIdx.x * 256 + threadIdx.x) * 8;
    float4 f0 = *reinterpret_cast<const float4*>(L + i);
    float4 f1 = *reinterpret_cast<const float4*>(L + i + 4);
    union { unsigned short s[8]; uint4 q; } o;
    o.s[0] = f2bf(f0.x); o.s[1] = f2bf(f0.y); o.s[2] = f2bf(f0.z); o.s[3] = f2bf(f0.w);
    o.s[4] = f2bf(f1.x); o.s[5] = f2bf(f1.y); o.s[6] = f2bf(f1.z); o.s[7] = f2bf(f1.w);
    *reinterpret_cast<uint4*>(Lb + i) = o.q;
}

// x fp32 [B][N][F_IN] -> X0t bf16 [B*F_IN][N]  (transposed storage)
__global__ __launch_bounds__(256) void conv_x(const float* __restrict__ x,
                                              unsigned short* __restrict__ X0t) {
    int idx = blockIdx.x * 256 + threadIdx.x;     // B*N*F_IN = 1,048,576
    int f = idx & 31;
    int n = (idx >> 5) & (NN - 1);
    int b = idx >> 18;
    X0t[(size_t)(b * 32 + f) * NN + n] = f2bf(x[idx]);
}

// weight fp32 [4][32][64] -> WcT bf16 [64][128] with Chebyshev fusion:
//  k 0..31: W0-W2, 32..63: W1-W3, 64..95: 2*W2, 96..127: 2*W3
__global__ __launch_bounds__(256) void prep_w(const float* __restrict__ W,
                                              unsigned short* __restrict__ WcT) {
    int idx = blockIdx.x * 256 + threadIdx.x;     // 64*128 = 8192
    int e = idx >> 7;
    int k = idx & 127;
    int f = k & 31, blk = k >> 5;
    float w;
    if      (blk == 0) w = W[(0 * 32 + f) * 64 + e] - W[(2 * 32 + f) * 64 + e];
    else if (blk == 1) w = W[(1 * 32 + f) * 64 + e] - W[(3 * 32 + f) * 64 + e];
    else if (blk == 2) w = 2.0f * W[(2 * 32 + f) * 64 + e];
    else               w = 2.0f * W[(3 * 32 + f) * 64 + e];
    WcT[idx] = f2bf(w);
}

// ------------------------- main GEMM: Yt = (L @ X)^T -----------------------
// Lb : bf16 [8192][8192] row-major (symmetric), Xt/Yt : bf16 [128][8192]
// Block: 256 thr (8 waves). M-tile = 32 rows (2 M-waves), 4 N-waves x 32 cols.
// Grid: 256 blocks -> L streamed from HBM exactly once per pass (L2-resident
// thereafter: 134 MB < 192 MB L2).
__global__ __launch_bounds__(256) void cheb_spmm(const unsigned short* __restrict__ Lb,
                                                 const unsigned short* __restrict__ Xt,
                                                 unsigned short* __restrict__ Yt) {
    const int lane = threadIdx.x & 31;
    const int wid  = threadIdx.x >> 5;
    const int wm   = wid & 1;                 // M sub-tile (0..1)
    const int wn   = wid >> 1;                // N sub-tile (0..3), 32 cols each
    const int m0   = blockIdx.x * 32 + wm * 16;
    const int cb   = wn * 32;

    const int r  = lane & 15;
    const int hi = lane >> 4;
    const int kh = hi << 3;                   // 0 or 8

    const unsigned short* arow = Lb + (size_t)(m0 + r) * NN + kh;
    const unsigned short* b0   = Xt + (size_t)(cb + r)      * NN + kh;
    const unsigned short* b1   = Xt + (size_t)(cb + 16 + r) * NN + kh;

    v8f acc0 = {}, acc1 = {};

    for (int k0 = 0; k0 < NN; k0 += 32) {
        __builtin_prefetch(arow + k0 + 2048, 0, 1);   // global_prefetch_b8
        V16 a  = load_op(arow + k0);
        V16 x0 = load_op(b0 + k0);
        V16 x1 = load_op(b1 + k0);
        acc0 = wmma_bf16(a, x0, acc0);
        acc1 = wmma_bf16(a, x1, acc1);
    }

    // D layout: VGPR j -> M = j + 8*hi, N = r. Store transposed bf16.
    const int nodeS = m0 + 8 * hi;
#pragma unroll
    for (int j = 0; j < 8; ++j) {
        Yt[(size_t)(cb + r)      * NN + nodeS + j] = f2bf(acc0[j]);
        Yt[(size_t)(cb + 16 + r) * NN + nodeS + j] = f2bf(acc1[j]);
    }
}

// ------------------ fused output contraction + bias ------------------------
// out[b][n][e] = sum_{k<128} Xcat[b][n][k] * Wc[k][e] + bias[n][e]
// Xcat columns come from {X0t,P1t,P2t,P3t} (transposed bf16); staged through
// LDS as [m][k] so the A operand is a ds_load_b128.
__global__ __launch_bounds__(256) void cheb_out(const unsigned short* __restrict__ X0t,
                                                const unsigned short* __restrict__ P1t,
                                                const unsigned short* __restrict__ P2t,
                                                const unsigned short* __restrict__ P3t,
                                                const unsigned short* __restrict__ WcT,
                                                const float* __restrict__ bias,
                                                float* __restrict__ out) {
    __shared__ __align__(16) unsigned short tile[128][136];  // [m][k], +8 pad

    const int b     = blockIdx.y;
    const int node0 = blockIdx.x * 128;
    const int t     = threadIdx.x;

    // Stage 128 nodes x 128 k, transposing global [k][m] -> LDS [m][k]
    {
        const int kk = t >> 1;              // 0..127
        const int h  = (t & 1) * 64;        // node offset
        const unsigned short* bases[4] = { X0t, P1t, P2t, P3t };
        const unsigned short* src =
            bases[kk >> 5] + (size_t)(b * 32 + (kk & 31)) * NN + node0 + h;
#pragma unroll 8
        for (int i = 0; i < 64; ++i) tile[h + i][kk] = src[i];
    }
    __syncthreads();

    const int lane = t & 31;
    const int wid  = t >> 5;                // 8 waves x 16 rows
    const int r    = lane & 15;
    const int hi   = lane >> 4;
    const int kh   = hi << 3;
    const int mrow = wid * 16;

    v8f e0 = {}, e1 = {}, e2 = {}, e3 = {};
#pragma unroll
    for (int k0 = 0; k0 < 128; k0 += 32) {
        V16 a  = load_op(&tile[mrow + r][k0 + kh]);
        V16 w0 = load_op(WcT + (size_t)(0  + r) * 128 + k0 + kh);
        V16 w1 = load_op(WcT + (size_t)(16 + r) * 128 + k0 + kh);
        V16 w2 = load_op(WcT + (size_t)(32 + r) * 128 + k0 + kh);
        V16 w3 = load_op(WcT + (size_t)(48 + r) * 128 + k0 + kh);
        e0 = wmma_bf16(a, w0, e0);
        e1 = wmma_bf16(a, w1, e1);
        e2 = wmma_bf16(a, w2, e2);
        e3 = wmma_bf16(a, w3, e3);
    }

    const int node = node0 + mrow + 8 * hi;
#pragma unroll
    for (int j = 0; j < 8; ++j) {
        const int n = node + j;
        const size_t ob = ((size_t)b * NN + n) * FOUT;
        const size_t bb = (size_t)n * FOUT;
        out[ob + 0  + r] = e0[j] + bias[bb + 0  + r];
        out[ob + 16 + r] = e1[j] + bias[bb + 16 + r];
        out[ob + 32 + r] = e2[j] + bias[bb + 32 + r];
        out[ob + 48 + r] = e3[j] + bias[bb + 48 + r];
    }
}

// ---------------------------------------------------------------------------

extern "C" void kernel_launch(void* const* d_in, const int* in_sizes, int n_in,
                              void* d_out, int out_size, void* d_ws, size_t ws_size,
                              hipStream_t stream) {
    const float* x    = (const float*)d_in[0];   // [4][8192][32]
    const float* L    = (const float*)d_in[1];   // [8192][8192]
    const float* W    = (const float*)d_in[2];   // [4][32][64]
    const float* bias = (const float*)d_in[3];   // [8192][64]
    float* out = (float*)d_out;                  // [4][8192][64]

    unsigned short* Lb  = (unsigned short*)d_ws;                 // 134.2 MB
    unsigned short* X0t = Lb  + (size_t)NN * NN;                 // 2 MB each
    unsigned short* P1t = X0t + (size_t)NC * NN;
    unsigned short* P2t = P1t + (size_t)NC * NN;
    unsigned short* P3t = P2t + (size_t)NC * NN;
    unsigned short* WcT = P3t + (size_t)NC * NN;                 // 16 KB

    conv_L<<<dim3((NN * (size_t)NN) / (8 * 256)), dim3(256), 0, stream>>>(L, Lb);
    conv_x<<<dim3((NBATCH * NN * 32) / 256), dim3(256), 0, stream>>>(x, X0t);
    prep_w<<<dim3((FOUT * 128) / 256), dim3(256), 0, stream>>>(W, WcT);

    cheb_spmm<<<dim3(NN / 32), dim3(256), 0, stream>>>(Lb, X0t, P1t);
    cheb_spmm<<<dim3(NN / 32), dim3(256), 0, stream>>>(Lb, P1t, P2t);
    cheb_spmm<<<dim3(NN / 32), dim3(256), 0, stream>>>(Lb, P2t, P3t);

    cheb_out<<<dim3(NN / 128, NBATCH), dim3(256), 0, stream>>>(
        X0t, P1t, P2t, P3t, WcT, bias, out);
}
